// MixtureOfExperts_81604378624494
// MI455X (gfx1250) — matrix operand order repaired and checked by reading, not compiled, not run on hardware
//
#include <hip/hip_runtime.h>
#include <hip/hip_bf16.h>

// ---------------------------------------------------------------------------
// MoE (top-2 of 8 experts), SwiGLU FFN, C=1024, H=2730 (padded 2752), N=4096
// Sparse grouped-GEMM formulation using v_wmma_f32_16x16x32_bf16.
// Each wave owns a 32x32 output macro-tile (2x2 WMMA tiles) for higher
// arithmetic intensity; weights pre-converted to bf16 (L2-resident, 134 MB).
// ---------------------------------------------------------------------------

#define N_TOK   4096
#define CDIM    1024
#define HDIM    2730
#define HPAD    2752        // multiple of 32 for WMMA K loop
#define NEXP    8
#define ROWCAP  8448        // 2*N + 8*31 (32-aligned segments), = 66 * 128

typedef __attribute__((ext_vector_type(16))) __bf16 v16bf;
typedef __attribute__((ext_vector_type(8)))  float  v8f;

union FragU { uint4 u[2]; v16bf v; };

// A fragment: row-major M x K source, 16x32 tile at (r0, c0), leading dim ld.
// Lane layout (CDNA5 ISA, 16-bit A 16x32): lanes 0-15 -> M=lane, K=0..7 &
// 16..23; lanes 16-31 -> M=lane-16, K=8..15 & 24..31.  Two aligned b128 loads.
__device__ inline v16bf load_frag_a(const __bf16* base, int r0, int c0, int ld) {
    int lane = threadIdx.x & 31;
    int half = lane >> 4;
    int m    = lane & 15;
    const __bf16* p = base + (size_t)(r0 + m) * ld + c0 + half * 8;
    FragU f;
    f.u[0] = *(const uint4*)(p);        // K = half*8 .. +7
    f.u[1] = *(const uint4*)(p + 16);   // K = 16 + half*8 .. +7
    return f.v;
}

// B fragment for D = A * W^T: W row-major [n-dim rows][k-dim cols], tile
// B[k][n] = W[n0+n][k0+k].  Lanes 0-15: N=lane, K=0..7 & 16..23; lanes 16-31: K=8..15 & 24..31.
__device__ inline v16bf load_frag_bT(const __bf16* base, int n0, int k0, int ld) {
    int lane = threadIdx.x & 31;
    int half = lane >> 4;
    int n    = lane & 15;
    const __bf16* p = base + (size_t)(n0 + n) * ld + k0 + half * 8;
    FragU f;
    f.u[0] = *(const uint4*)(p);
    f.u[1] = *(const uint4*)(p + 16);
    return f.v;
}

__device__ inline v8f wmma_bf16(v16bf a, v16bf b, v8f c) {
    return __builtin_amdgcn_wmma_f32_16x16x32_bf16(
        false, a, false, b, (short)0, c, false, false);
}

// ------------------------------ init ---------------------------------------
__global__ void moe_init(int* counts, int* cursors) {
    if (threadIdx.x < NEXP) { counts[threadIdx.x] = 0; cursors[threadIdx.x] = 0; }
}

// --------------------------- weight convert --------------------------------
// dst[e][hp][c] = (hp < H) ? src[e][hp][c] : 0     (w1, w2)
__global__ void conv_pad_rows(const float* __restrict__ src, __bf16* __restrict__ dst) {
    size_t idx = (size_t)blockIdx.x * 256 + threadIdx.x;
    size_t total = (size_t)NEXP * HPAD * CDIM;
    if (idx >= total) return;
    int c = (int)(idx % CDIM);
    size_t t = idx / CDIM;
    int h = (int)(t % HPAD);
    int e = (int)(t / HPAD);
    float v = (h < HDIM) ? src[((size_t)e * HDIM + h) * CDIM + c] : 0.0f;
    dst[idx] = (__bf16)v;
}

// dst[e][c][hp] = (hp < H) ? src[e][c][hp] : 0     (w3)
__global__ void conv_pad_cols(const float* __restrict__ src, __bf16* __restrict__ dst) {
    size_t idx = (size_t)blockIdx.x * 256 + threadIdx.x;
    size_t total = (size_t)NEXP * CDIM * HPAD;
    if (idx >= total) return;
    int hp = (int)(idx % HPAD);
    size_t t = idx / HPAD;
    int c = (int)(t % CDIM);
    int e = (int)(t / CDIM);
    float v = (hp < HDIM) ? src[((size_t)e * CDIM + c) * HDIM + hp] : 0.0f;
    dst[idx] = (__bf16)v;
}

// ------------------------------ router -------------------------------------
// One wave (32 lanes) per token: 8 dot products over C, shuffle reduce,
// top-2 + softmax over the two selected logits.
__global__ void router(const float* __restrict__ x, const float* __restrict__ gate_w,
                       int* __restrict__ topIdx, float* __restrict__ topW,
                       int* __restrict__ counts) {
    int token = blockIdx.x * 8 + (threadIdx.x >> 5);
    int lane  = threadIdx.x & 31;
    if (token >= N_TOK) return;
    const float* xr = x + (size_t)token * CDIM;
    float acc[NEXP];
#pragma unroll
    for (int e = 0; e < NEXP; ++e) acc[e] = 0.0f;
    for (int c = lane; c < CDIM; c += 32) {
        float xv = xr[c];
#pragma unroll
        for (int e = 0; e < NEXP; ++e) acc[e] += xv * gate_w[e * CDIM + c];
    }
#pragma unroll
    for (int off = 16; off > 0; off >>= 1) {
#pragma unroll
        for (int e = 0; e < NEXP; ++e) acc[e] += __shfl_xor(acc[e], off, 32);
    }
    // top-2 (first occurrence wins ties, matching top_k)
    int i0 = 0; float v0 = acc[0];
#pragma unroll
    for (int e = 1; e < NEXP; ++e) if (acc[e] > v0) { v0 = acc[e]; i0 = e; }
    int i1 = (i0 == 0) ? 1 : 0; float v1 = acc[i1];
#pragma unroll
    for (int e = 0; e < NEXP; ++e)
        if (e != i0 && acc[e] > v1) { v1 = acc[e]; i1 = e; }
    float e1  = __expf(v1 - v0);             // v0 >= v1
    float inv = 1.0f / (1.0f + e1);
    if (lane == 0) {
        topIdx[2 * token + 0] = i0;
        topIdx[2 * token + 1] = i1;
        topW[2 * token + 0] = inv;
        topW[2 * token + 1] = e1 * inv;
        atomicAdd(&counts[i0], 1);
        atomicAdd(&counts[i1], 1);
    }
}

// ------------------------------ scan ---------------------------------------
__global__ void scan_offsets(const int* __restrict__ counts, int* __restrict__ offsets,
                             int* __restrict__ cursors) {
    if (threadIdx.x == 0) {
        int off = 0;
        for (int e = 0; e < NEXP; ++e) {
            offsets[e] = off;
            off += (counts[e] + 31) & ~31;   // 32-align each expert segment
        }
        offsets[NEXP] = off;
    }
    if (threadIdx.x < NEXP) cursors[threadIdx.x] = 0;
}

// ------------------------------ scatter ------------------------------------
__global__ void scatter_slots(const int* __restrict__ topIdx, const int* __restrict__ offsets,
                              int* __restrict__ cursors, int* __restrict__ tokenSlot) {
    int n = blockIdx.x * blockDim.x + threadIdx.x;
    if (n >= N_TOK) return;
#pragma unroll
    for (int k = 0; k < 2; ++k) {
        int e   = topIdx[2 * n + k];
        int pos = atomicAdd(&cursors[e], 1);
        tokenSlot[2 * n + k] = offsets[e] + pos;
    }
}

// ------------------------------ gather -------------------------------------
// Copy token rows (fp32 -> bf16) into their assigned expert-segment rows.
__global__ void gather_rows(const float* __restrict__ x, const int* __restrict__ tokenSlot,
                            __bf16* __restrict__ xg) {
    int a   = blockIdx.x;           // assignment id 0..2N-1
    int n   = a >> 1;
    int row = tokenSlot[a];
    const float* src = x + (size_t)n * CDIM;
    __bf16* dst = xg + (size_t)row * CDIM;
    for (int c = threadIdx.x; c < CDIM; c += 256) dst[c] = (__bf16)src[c];
}

// --------------------------- FFN pass 1 (WMMA) -----------------------------
// act[row, h] = silu(xg . w1^T) * (xg . w2^T), per-expert segments.
// Block = 8 waves = 4 row-groups x 2 h-groups; each wave owns a 32x32
// macro-tile (2x2 WMMA tiles, both w1 and w2 -> 8 WMMAs per K chunk,
// 6 fragment loads).  Block covers 128 rows x 64 h.
__global__ void ffn1(const __bf16* __restrict__ xg,
                     const __bf16* __restrict__ w1b,
                     const __bf16* __restrict__ w2b,
                     __bf16* __restrict__ act,
                     const int* __restrict__ offsets) {
    int wave = threadIdx.x >> 5;
    int r0 = blockIdx.x * 128 + (wave >> 1) * 32;
    int h0 = blockIdx.y * 64  + (wave & 1) * 32;
    int totalRows = offsets[NEXP];
    if (r0 >= totalRows) return;           // wave-uniform (segments 32-aligned)
    int e = 0;
#pragma unroll
    for (int i = 1; i < NEXP; ++i) if (offsets[i] <= r0) e = i;

    const __bf16* W1 = w1b + (size_t)e * HPAD * CDIM;
    const __bf16* W2 = w2b + (size_t)e * HPAD * CDIM;

    v8f acc1[2][2] = {};                   // [row tile][h tile], silu input
    v8f acc2[2][2] = {};                   // gate input
    int lane = threadIdx.x & 31;
    for (int k = 0; k < CDIM; k += 32) {
        // speculative prefetch of the next weight chunk (L2-resident bf16)
        __builtin_prefetch(W1 + (size_t)(h0 + (lane & 31)) * CDIM + k + 32, 0, 0);
        __builtin_prefetch(W2 + (size_t)(h0 + (lane & 31)) * CDIM + k + 32, 0, 0);
        v16bf a0 = load_frag_a(xg, r0,      k, CDIM);
        v16bf a1 = load_frag_a(xg, r0 + 16, k, CDIM);
        v16bf b10 = load_frag_bT(W1, h0,      k, CDIM);
        v16bf b11 = load_frag_bT(W1, h0 + 16, k, CDIM);
        v16bf b20 = load_frag_bT(W2, h0,      k, CDIM);
        v16bf b21 = load_frag_bT(W2, h0 + 16, k, CDIM);
        acc1[0][0] = wmma_bf16(a0, b10, acc1[0][0]);
        acc1[0][1] = wmma_bf16(a0, b11, acc1[0][1]);
        acc1[1][0] = wmma_bf16(a1, b10, acc1[1][0]);
        acc1[1][1] = wmma_bf16(a1, b11, acc1[1][1]);
        acc2[0][0] = wmma_bf16(a0, b20, acc2[0][0]);
        acc2[0][1] = wmma_bf16(a0, b21, acc2[0][1]);
        acc2[1][0] = wmma_bf16(a1, b20, acc2[1][0]);
        acc2[1][1] = wmma_bf16(a1, b21, acc2[1][1]);
    }
    int n = lane & 15, half = lane >> 4;
#pragma unroll
    for (int rt = 0; rt < 2; ++rt) {
#pragma unroll
        for (int ht = 0; ht < 2; ++ht) {
#pragma unroll
            for (int r = 0; r < 8; ++r) {
                float s1 = acc1[rt][ht][r], s2 = acc2[rt][ht][r];
                float g  = s1 / (1.0f + __expf(-s1));     // silu
                act[(size_t)(r0 + rt * 16 + r + 8 * half) * HPAD
                    + h0 + ht * 16 + n] = (__bf16)(g * s2);
            }
        }
    }
}

// --------------------------- FFN pass 2 (WMMA) -----------------------------
// rowOut[row, c] = act . w3^T  (w3b row-major [C][HPAD], zero-padded cols).
// Block = 8 waves = 4 row-groups x 2 c-groups; wave owns 32x32 macro-tile
// (4 WMMAs per K chunk, 4 fragment loads).  Block covers 128 rows x 64 c.
__global__ void ffn2(const __bf16* __restrict__ act,
                     const __bf16* __restrict__ w3b,
                     float* __restrict__ rowOut,
                     const int* __restrict__ offsets) {
    int wave = threadIdx.x >> 5;
    int r0 = blockIdx.x * 128 + (wave >> 1) * 32;
    int c0 = blockIdx.y * 64  + (wave & 1) * 32;
    int totalRows = offsets[NEXP];
    if (r0 >= totalRows) return;           // wave-uniform (segments 32-aligned)
    int e = 0;
#pragma unroll
    for (int i = 1; i < NEXP; ++i) if (offsets[i] <= r0) e = i;

    const __bf16* W3 = w3b + (size_t)e * CDIM * HPAD;

    v8f acc[2][2] = {};
    int lane = threadIdx.x & 31;
    for (int k = 0; k < HPAD; k += 32) {
        __builtin_prefetch(W3 + (size_t)(c0 + (lane & 31)) * HPAD + k + 32, 0, 0);
        v16bf a0 = load_frag_a(act, r0,      k, HPAD);
        v16bf a1 = load_frag_a(act, r0 + 16, k, HPAD);
        v16bf b0 = load_frag_bT(W3, c0,      k, HPAD);
        v16bf b1 = load_frag_bT(W3, c0 + 16, k, HPAD);
        acc[0][0] = wmma_bf16(a0, b0, acc[0][0]);
        acc[0][1] = wmma_bf16(a0, b1, acc[0][1]);
        acc[1][0] = wmma_bf16(a1, b0, acc[1][0]);
        acc[1][1] = wmma_bf16(a1, b1, acc[1][1]);
    }
    int n = lane & 15, half = lane >> 4;
#pragma unroll
    for (int rt = 0; rt < 2; ++rt)
#pragma unroll
        for (int ct = 0; ct < 2; ++ct)
#pragma unroll
            for (int r = 0; r < 8; ++r)
                rowOut[(size_t)(r0 + rt * 16 + r + 8 * half) * CDIM
                       + c0 + ct * 16 + n] = acc[rt][ct][r];
}

// ------------------------------ combine ------------------------------------
// Deterministic weighted gather of each token's two expert rows.
__global__ void combine(const float* __restrict__ rowOut, const int* __restrict__ tokenSlot,
                        const float* __restrict__ topW, float* __restrict__ out) {
    int n = blockIdx.x;
    int s0 = tokenSlot[2 * n + 0], s1 = tokenSlot[2 * n + 1];
    float w0 = topW[2 * n + 0], w1 = topW[2 * n + 1];
    const float* r0 = rowOut + (size_t)s0 * CDIM;
    const float* r1 = rowOut + (size_t)s1 * CDIM;
    float* dst = out + (size_t)n * CDIM;
    for (int c = threadIdx.x; c < CDIM; c += 256)
        dst[c] = w0 * r0[c] + w1 * r1[c];
}

// ---------------------------------------------------------------------------
extern "C" void kernel_launch(void* const* d_in, const int* in_sizes, int n_in,
                              void* d_out, int out_size, void* d_ws, size_t ws_size,
                              hipStream_t stream) {
    const float* x      = (const float*)d_in[0];   // [2,2048,1024]
    const float* gate_w = (const float*)d_in[1];   // [8,1024]
    const float* w1     = (const float*)d_in[2];   // [8,2730,1024]
    const float* w2     = (const float*)d_in[3];   // [8,2730,1024]
    const float* w3     = (const float*)d_in[4];   // [8,1024,2730]
    float* out          = (float*)d_out;

    // ---- workspace carve-up (256B aligned sections) ----
    char* p = (char*)d_ws;
    auto alloc = [&](size_t bytes) -> char* {
        char* r = p;
        p += (bytes + 255) & ~(size_t)255;
        return r;
    };
    int*    counts    = (int*)   alloc(NEXP * 4);
    int*    cursors   = (int*)   alloc(NEXP * 4);
    int*    offsets   = (int*)   alloc((NEXP + 1) * 4);
    int*    topIdx    = (int*)   alloc((size_t)N_TOK * 2 * 4);
    float*  topW      = (float*) alloc((size_t)N_TOK * 2 * 4);
    int*    tokenSlot = (int*)   alloc((size_t)N_TOK * 2 * 4);
    __bf16* xg        = (__bf16*)alloc((size_t)ROWCAP * CDIM * 2);
    __bf16* w1b       = (__bf16*)alloc((size_t)NEXP * HPAD * CDIM * 2);
    __bf16* w2b       = (__bf16*)alloc((size_t)NEXP * HPAD * CDIM * 2);
    __bf16* w3b       = (__bf16*)alloc((size_t)NEXP * CDIM * HPAD * 2);
    __bf16* act       = (__bf16*)alloc((size_t)ROWCAP * HPAD * 2);
    float*  rowOut    = (float*) alloc((size_t)ROWCAP * CDIM * 4);
    (void)ws_size; (void)in_sizes; (void)n_in; (void)out_size;

    // ---- phase 0: init + weight conversion (fp32 -> L2-resident bf16) ----
    moe_init<<<1, 64, 0, stream>>>(counts, cursors);
    {
        size_t total = (size_t)NEXP * HPAD * CDIM;
        int blocks = (int)((total + 255) / 256);
        conv_pad_rows<<<blocks, 256, 0, stream>>>(w1, w1b);
        conv_pad_rows<<<blocks, 256, 0, stream>>>(w2, w2b);
        conv_pad_cols<<<blocks, 256, 0, stream>>>(w3, w3b);
    }

    // ---- phase 1: router / segment build / gather ----
    router<<<N_TOK / 8, 256, 0, stream>>>(x, gate_w, topIdx, topW, counts);
    scan_offsets<<<1, 32, 0, stream>>>(counts, offsets, cursors);
    scatter_slots<<<(N_TOK + 255) / 256, 256, 0, stream>>>(topIdx, offsets, cursors, tokenSlot);
    gather_rows<<<N_TOK * 2, 256, 0, stream>>>(x, tokenSlot, xg);

    // ---- phase 2: grouped expert FFN via WMMA (32x32 macro-tiles/wave) ----
    dim3 g1(ROWCAP / 128, HPAD / 64);   // 66 x 43
    ffn1<<<g1, 256, 0, stream>>>(xg, w1b, w2b, act, offsets);

    dim3 g2(ROWCAP / 128, CDIM / 64);   // 66 x 16
    ffn2<<<g2, 256, 0, stream>>>(act, w3b, rowOut, offsets);

    // ---- phase 3: deterministic weighted combine ----
    combine<<<N_TOK, 256, 0, stream>>>(rowOut, tokenSlot, topW, out);
}
